// SelfAttention_37881611551174
// MI455X (gfx1250) — matrix operand order repaired and checked by reading, not compiled, not run on hardware
//
#include <hip/hip_runtime.h>

#define B_ 8
#define S_ 2048
#define D_ 512

typedef __attribute__((ext_vector_type(16))) __bf16 v16bf;
typedef __attribute__((ext_vector_type(8)))  float  v8f;
typedef __attribute__((ext_vector_type(4)))  unsigned int u32x4;
typedef __attribute__((ext_vector_type(8)))  int          i32x8;
typedef __attribute__((ext_vector_type(4)))  int          i32x4;

union ABFrag { v16bf v; uint4 q[2]; };

__device__ inline unsigned short f2bf(float f) {
  unsigned u = __float_as_uint(f);
  u += 0x7FFFu + ((u >> 16) & 1u);   // round-to-nearest-even
  return (unsigned short)(u >> 16);
}

// ---- x (fp32) -> bf16 workspace copy; 8 elements / thread ----
__global__ void cvt_bf16_kernel(const float* __restrict__ x,
                                unsigned short* __restrict__ xb) {
  const int i = blockIdx.x * blockDim.x + threadIdx.x;
  const float4* src = (const float4*)x;
  float4 f0 = src[2 * i];
  float4 f1 = src[2 * i + 1];
  uint4 o;
  o.x = (unsigned)f2bf(f0.x) | ((unsigned)f2bf(f0.y) << 16);
  o.y = (unsigned)f2bf(f0.z) | ((unsigned)f2bf(f0.w) << 16);
  o.z = (unsigned)f2bf(f1.x) | ((unsigned)f2bf(f1.y) << 16);
  o.w = (unsigned)f2bf(f1.z) | ((unsigned)f2bf(f1.w) << 16);
  ((uint4*)xb)[i] = o;
}

__global__ void zero_kernel(float* __restrict__ p, int n) {
  const int i = blockIdx.x * blockDim.x + threadIdx.x;
  if (i < n) p[i] = 0.0f;
}

// ---- main kernel: one block = (b, 16-row q tile); 8 waves cover k=2048 ----
__launch_bounds__(256)
__global__ void attn_scores_kernel(const unsigned short* __restrict__ xb,
                                   const int* __restrict__ mask,
                                   float* __restrict__ p_out,
                                   float* __restrict__ colsum) {
  const int tid  = threadIdx.x;
  const int lane = tid & 31;
  const int w    = tid >> 5;          // wave id 0..7
  const int bid  = blockIdx.x;
  const int b    = bid >> 7;          // S/16 = 128 q-tiles per batch
  const int q0   = (bid & 127) << 4;

  const int m  = lane & 15;           // row-in-tile (A) / col-in-tile (B,C)
  const int kh = lane >> 4;           // K-half selector per ISA layout
  const int rbase = kh * 8;           // C rows owned by this half

  const unsigned short* xbB = xb + (size_t)b * S_ * D_;
  const int* maskB = mask + ((size_t)b * S_ + q0) * S_;  // 16 rows x 8 KB

  __shared__ unsigned short sA[16 * D_];   // 16 KB staged A tile
  __shared__ float rs[8][16];

  // ---- prefetch the block's mask region (128 KB) into cache while we GEMM ----
  for (int off = tid; off < (16 * S_) / 32; off += 256)
    __builtin_prefetch(maskB + off * 32, 0, 1);

  // ---- stage A tile (contiguous 16 KB of bf16 x) into LDS via the TDM ----
#if __has_builtin(__builtin_amdgcn_tensor_load_to_lds)
  if (tid < 32) {                     // wave 0 drives the Tensor Data Mover
    const unsigned long long ga = (unsigned long long)(uintptr_t)(xbB + (size_t)q0 * D_);
    const unsigned lds = (unsigned)(uintptr_t)(&sA[0]);
    u32x4 g0;
    g0[0] = 1u;                                            // count=1, user D#
    g0[1] = lds;                                           // lds_addr
    g0[2] = (unsigned)(ga & 0xFFFFFFFFu);                  // global_addr[31:0]
    g0[3] = (unsigned)((ga >> 32) & 0x01FFFFFFu) | 0x80000000u; // addr[56:32] | type=2
    i32x8 g1;
    g1[0] = 0x00030000;               // wg_mask=0, data_size=3 (8-byte units)
    g1[1] = (int)(2048u << 16);       // tensor_dim0 = 2048 (low 16 @ [31:16])
    g1[2] = (int)(1u << 16);          // tensor_dim0 hi=0 | tensor_dim1 = 1
    g1[3] = (int)(2048u << 16);       // tensor_dim1 hi=0 | tile_dim0 = 2048
    g1[4] = 1;                        // tile_dim1 = 1, tile_dim2 = 0
    g1[5] = 2048;                     // tensor_dim0_stride low 32
    g1[6] = (int)(2048u << 16);       // stride0 hi=0 | tensor_dim1_stride lo16
    g1[7] = 0;
    i32x4 gz4 = {0, 0, 0, 0};
    i32x8 gz8 = {0, 0, 0, 0, 0, 0, 0, 0};
    __builtin_amdgcn_tensor_load_to_lds(g0, g1, gz4, gz4, gz8, 0);
    __builtin_amdgcn_s_wait_tensorcnt(0);
  }
#else
  for (int e = tid; e < (16 * D_) / 8; e += 256)           // cooperative copy
    ((uint4*)sA)[e] = ((const uint4*)(xbB + (size_t)q0 * D_))[e];
#endif
  __syncthreads();

  v8f acc[16];
  const v8f vz = {0.f, 0.f, 0.f, 0.f, 0.f, 0.f, 0.f, 0.f};
#pragma unroll
  for (int t = 0; t < 16; ++t) acc[t] = vz;

  // ---- Gram GEMM: scores_tile = x_q (16xD) * x_k^T (Dx16), K-step 32 ----
  const uint4* sAq = (const uint4*)sA;
  const int arow = m * D_ + kh * 8;                        // ushort offset in sA
  for (int ds = 0; ds < D_; ds += 32) {
    ABFrag a;                          // A 16x32 bf16 from LDS (ds_load_b128)
    a.q[0] = sAq[(arow + ds) >> 3];
    a.q[1] = sAq[(arow + ds + 16) >> 3];
#pragma unroll
    for (int t = 0; t < 16; ++t) {
      const int k0 = (((t << 3) + w) << 4);
      const unsigned short* brow = xbB + (size_t)(k0 + m) * D_ + ds + kh * 8;
      ABFrag bf;                       // B 32x16 bf16: two 8-halfword chunks at +0 and +16
      bf.q[0] = *(const uint4*)(brow);
      bf.q[1] = *(const uint4*)(brow + 16);
      acc[t] = __builtin_amdgcn_wmma_f32_16x16x32_bf16(
          false, a.v, false, bf.v, (short)0, acc[t], false, false);
    }
  }

  // ---- exp / mask / unnormalized softmax, per-row partial sums ----
  const float invSqrtD = 0.044194173824159216f;  // 1/sqrt(512)
  float rsum[8];
#pragma unroll
  for (int v = 0; v < 8; ++v) rsum[v] = 0.f;

#pragma unroll
  for (int t = 0; t < 16; ++t) {
    const int col = (((t << 3) + w) << 4) + m;
#pragma unroll
    for (int v = 0; v < 8; ++v) {
      const int mk = maskB[(size_t)(rbase + v) * S_ + col];
      float p = 0.f;
      if (mk != 0) p = __expf(__expf(acc[t][v]) * invSqrtD);
      acc[t][v] = p;
      rsum[v] += p;
    }
  }

  // reduce row sums across the 16 lanes of each half (cols 0..15)
#pragma unroll
  for (int off = 1; off < 16; off <<= 1) {
#pragma unroll
    for (int v = 0; v < 8; ++v) rsum[v] += __shfl_xor(rsum[v], off, 32);
  }

  // combine across the 8 waves (each owns 256 of 2048 k-columns)
  if (lane == 0) {
#pragma unroll
    for (int v = 0; v < 8; ++v) rs[w][v] = rsum[v];
  } else if (lane == 16) {
#pragma unroll
    for (int v = 0; v < 8; ++v) rs[w][8 + v] = rsum[v];
  }
  __syncthreads();

  float recip[8];
#pragma unroll
  for (int v = 0; v < 8; ++v) {
    const int row = rbase + v;
    float s = 0.f;
#pragma unroll
    for (int ww = 0; ww < 8; ++ww) s += rs[ww][row];
    recip[v] = 1.0f / s;
  }

  // ---- normalize, store p_attn, fuse column sums for the output GEMV ----
  float* poutB = p_out + (size_t)b * S_ * S_;
  float* csB   = colsum + (size_t)b * S_;
#pragma unroll
  for (int t = 0; t < 16; ++t) {
    const int col = (((t << 3) + w) << 4) + m;
    float cs = 0.f;
#pragma unroll
    for (int v = 0; v < 8; ++v) {
      const int row = q0 + rbase + v;
      const float p = acc[t][v] * recip[v];
      poutB[(size_t)row * S_ + col] = p;
      cs += p;
    }
    cs += __shfl_xor(cs, 16, 32);     // rows 0-7 half + rows 8-15 half
    if (lane < 16) unsafeAtomicAdd(&csB[col], cs);
  }
}

// ---- out[b,d] = sum_k colsum[b,k] * x[b,k,d]  (fp32 x for accuracy) ----
__global__ void out_kernel(const float* __restrict__ x,
                           const float* __restrict__ colsum,
                           float* __restrict__ out) {
  const int b = blockIdx.x;
  const float* xB = x + (size_t)b * S_ * D_;
  const float* cs = colsum + (size_t)b * S_;
  for (int d = threadIdx.x; d < D_; d += blockDim.x) {
    float acc = 0.f;
    for (int k = 0; k < S_; ++k) acc = fmaf(cs[k], xB[(size_t)k * D_ + d], acc);
    out[b * D_ + d] = acc;
  }
}

extern "C" void kernel_launch(void* const* d_in, const int* in_sizes, int n_in,
                              void* d_out, int out_size, void* d_ws, size_t ws_size,
                              hipStream_t stream) {
  const float* x    = (const float*)d_in[0];
  const int*   mask = (const int*)d_in[1];

  float* out0  = (float*)d_out;                  // (B,D) flattened first
  float* p_out = out0 + (size_t)B_ * D_;         // then p_attn (B,S,S)

  unsigned short* xb = (unsigned short*)d_ws;    // bf16 x copy: 16 MB
  float* colsum = (float*)((char*)d_ws + (size_t)B_ * S_ * D_ * sizeof(unsigned short));

  // 1) fp32 -> bf16 (8 elems / thread)
  const int nvec = (B_ * S_ * D_) / 8;
  cvt_bf16_kernel<<<nvec / 256, 256, 0, stream>>>(x, xb);

  // 2) zero the column-sum accumulator (deterministic each call)
  zero_kernel<<<(B_ * S_ + 255) / 256, 256, 0, stream>>>(colsum, B_ * S_);

  // 3) WMMA Gram + softmax + p_attn store + fused column sums
  attn_scores_kernel<<<B_ * (S_ / 16), 256, 0, stream>>>(xb, mask, p_out, colsum);

  // 4) tiny output contraction in fp32
  out_kernel<<<B_, 256, 0, stream>>>(x, colsum, out0);
}
